// SoftSOM_46643344834799
// MI455X (gfx1250) — matrix operand order repaired and checked by reading, not compiled, not run on hardware
//
#include <hip/hip_runtime.h>

typedef __attribute__((ext_vector_type(16))) __bf16 bf16x16;
typedef __attribute__((ext_vector_type(8)))  __bf16 bf16x8;
typedef __attribute__((ext_vector_type(4)))  __bf16 bf16x4;
typedef __attribute__((ext_vector_type(8)))  float  f32x8;
typedef __attribute__((ext_vector_type(4)))  float  f32x4;

#define DIM 128
#define NC 256
#define WAVES 8
#define ROWS_PER_WAVE 16
#define ROWS_PER_BLOCK (WAVES * ROWS_PER_WAVE)
#define STRIDE_CM 136   /* c-major row stride in bf16 (128 + 8 pad) -> 272B, 16B aligned */
#define STRIDE_DM 264   /* d-major row stride in bf16 (256 + 8 pad) -> 528B, 16B aligned */

__device__ __forceinline__ float sigmoid_f(float v) {
    return 1.0f / (1.0f + __expf(-v));
}

__device__ __forceinline__ bf16x16 join8(bf16x8 lo, bf16x8 hi) {
    bf16x16 a;
    #pragma unroll
    for (int e = 0; e < 8; ++e) { a[e] = lo[e]; a[e + 8] = hi[e]; }
    return a;
}

__global__ __launch_bounds__(256) void softsom_kernel(
    const float* __restrict__ x,
    const float* __restrict__ protos,
    const float* __restrict__ grid,
    const float* __restrict__ temp_raw,
    const float* __restrict__ gate_logits,
    float* __restrict__ out_blended,
    float* __restrict__ out_w)
{
    // One LDS matrix buffer, reused 3x: protos (c-major) -> grid (c-major) -> protos (d-major)
    __shared__ __align__(16) __bf16 smat[NC * STRIDE_CM];   // 69632 B (>= 128*264*2 = 67584 B)
    __shared__ float p2s[NC];
    __shared__ float g2s[NC];
    __shared__ float gateS[NC];

    const int tid  = threadIdx.x;
    const int lane = tid & 31;
    const int wave = tid >> 5;
    const int hl   = lane >> 4;      // 0: lanes 0-15, 1: lanes 16-31
    const int ln16 = lane & 15;

    const int row0  = blockIdx.x * ROWS_PER_BLOCK + wave * ROWS_PER_WAVE;
    const int myrow = row0 + ln16;   // x-row owned by this lane (shared with partner lane ^16)

    // ---------------- Phase 0: gate, prototypes -> LDS (c-major bf16), |p|^2 ----------------
    gateS[tid] = sigmoid_f(gate_logits[tid]);
    {
        const f32x4* src = (const f32x4*)(protos + (size_t)tid * DIM);
        float s = 0.f;
        #pragma unroll
        for (int i = 0; i < DIM / 4; ++i) {
            f32x4 v = src[i];
            s += v[0]*v[0] + v[1]*v[1] + v[2]*v[2] + v[3]*v[3];
            bf16x4 b;
            #pragma unroll
            for (int e = 0; e < 4; ++e) b[e] = (__bf16)v[e];
            *(bf16x4*)(smat + tid * STRIDE_CM + i * 4) = b;
        }
        p2s[tid] = s;
    }
    __syncthreads();

    // ---------------- Load x B-fragments (bf16) + own-half |x|^2 ----------------
    // B frag k, element j  <->  x[myrow][32k + 16*hl + j]  (16 contiguous f32 per lane)
    bf16x16 xf[4];
    float x2p = 0.f;
    {
        const float* xrow = x + (size_t)myrow * DIM + hl * 16;
        #pragma unroll
        for (int k = 0; k < 4; ++k) {
            const f32x4* p4 = (const f32x4*)(xrow + 32 * k);
            #pragma unroll
            for (int q = 0; q < 4; ++q) {
                f32x4 v = p4[q];
                #pragma unroll
                for (int e = 0; e < 4; ++e) {
                    float f = v[e];
                    x2p += f * f;
                    xf[k][q * 4 + e] = (__bf16)f;
                }
            }
        }
    }
    const float x2 = x2p + __shfl_xor(x2p, 16, 32);

    // ---------------- Pass 1: dotP^T tiles = P_tile(16xD) * x^T ----------------
    // A frag: row c = ct*16 + ln16, elem j at d = 32k + 16*hl + j
    // D tile: lane -> (x-row = ln16, c offset r + 8*hl)
    f32x8 wreg[16];
    {
        const __bf16* abase = smat + ln16 * STRIDE_CM + 16 * hl;
        #pragma unroll
        for (int ct = 0; ct < 16; ++ct) {
            const __bf16* rp = abase + ct * 16 * STRIDE_CM;
            // issue all 8 fragment loads up-front so WMMAs can overlap LDS latency
            bf16x16 a[4];
            #pragma unroll
            for (int k = 0; k < 4; ++k) {
                bf16x8 lo = *(const bf16x8*)(rp + 32 * k);
                bf16x8 hi = *(const bf16x8*)(rp + 32 * k + 8);
                a[k] = join8(lo, hi);
            }
            f32x8 acc = {0.f, 0.f, 0.f, 0.f, 0.f, 0.f, 0.f, 0.f};
            #pragma unroll
            for (int k = 0; k < 4; ++k)
                acc = __builtin_amdgcn_wmma_f32_16x16x32_bf16(false, a[k], false, xf[k],
                                                              (short)0, acc, false, false);
            wreg[ct] = acc;   // holds dotP for c = ct*16 + 8*hl + r
        }
    }
    __syncthreads();

    // ---------------- Phase 0b: grid_pos -> LDS (c-major bf16), |g|^2 ----------------
    {
        const f32x4* src = (const f32x4*)(grid + (size_t)tid * DIM);
        float s = 0.f;
        #pragma unroll
        for (int i = 0; i < DIM / 4; ++i) {
            f32x4 v = src[i];
            s += v[0]*v[0] + v[1]*v[1] + v[2]*v[2] + v[3]*v[3];
            bf16x4 b;
            #pragma unroll
            for (int e = 0; e < 4; ++e) b[e] = (__bf16)v[e];
            *(bf16x4*)(smat + tid * STRIDE_CM + i * 4) = b;
        }
        g2s[tid] = s;
    }
    __syncthreads();

    // ---------------- Pass 2: dotG^T + distances + unnormalized w + row sums ----------------
    const float temperature = sigmoid_f(temp_raw[0]) * (1.0f - 1e-3f) + 1e-3f;
    const float invT = 1.0f / temperature;

    float ps = 0.f, gs = 0.f;
    {
        const __bf16* abase = smat + ln16 * STRIDE_CM + 16 * hl;
        #pragma unroll
        for (int ct = 0; ct < 16; ++ct) {
            const __bf16* rp = abase + ct * 16 * STRIDE_CM;
            bf16x16 a[4];
            #pragma unroll
            for (int k = 0; k < 4; ++k) {
                bf16x8 lo = *(const bf16x8*)(rp + 32 * k);
                bf16x8 hi = *(const bf16x8*)(rp + 32 * k + 8);
                a[k] = join8(lo, hi);
            }
            f32x8 acc = {0.f, 0.f, 0.f, 0.f, 0.f, 0.f, 0.f, 0.f};
            #pragma unroll
            for (int k = 0; k < 4; ++k)
                acc = __builtin_amdgcn_wmma_f32_16x16x32_bf16(false, a[k], false, xf[k],
                                                              (short)0, acc, false, false);

            const int cbase = ct * 16 + 8 * hl;
            f32x4 p2a = *(const f32x4*)(p2s + cbase), p2b = *(const f32x4*)(p2s + cbase + 4);
            f32x4 g2a = *(const f32x4*)(g2s + cbase), g2b = *(const f32x4*)(g2s + cbase + 4);
            f32x4 gta = *(const f32x4*)(gateS + cbase), gtb = *(const f32x4*)(gateS + cbase + 4);
            #pragma unroll
            for (int r = 0; r < 8; ++r) {
                float p2v = (r < 4) ? p2a[r & 3] : p2b[r & 3];
                float g2v = (r < 4) ? g2a[r & 3] : g2b[r & 3];
                float gv  = (r < 4) ? gta[r & 3] : gtb[r & 3];
                float d2p = x2 + p2v - 2.0f * wreg[ct][r];
                float d2g = x2 + g2v - 2.0f * acc[r];
                float dp = __builtin_sqrtf(fmaxf(d2p, 1e-12f));
                float dg = __builtin_sqrtf(fmaxf(d2g, 1e-12f));
                float wv = __expf(-(dp + dg) * invT);
                wreg[ct][r] = wv;
                ps += wv;
                gs += wv * gv;
            }
        }
    }
    const float s1    = ps + __shfl_xor(ps, 16, 32) + 1e-8f;
    const float gsum  = gs + __shfl_xor(gs, 16, 32);
    const float s2    = gsum / s1 + 1e-8f;
    const float scale = 1.0f / (s1 * s2);

    // ---------------- Finalize w, write to global (32B contiguous per lane) ----------------
    {
        float* wrow = out_w + (size_t)myrow * NC + 8 * hl;
        #pragma unroll
        for (int ct = 0; ct < 16; ++ct) {
            const int cbase = ct * 16 + 8 * hl;
            f32x4 gta = *(const f32x4*)(gateS + cbase), gtb = *(const f32x4*)(gateS + cbase + 4);
            f32x4 oa, ob;
            #pragma unroll
            for (int r = 0; r < 4; ++r) {
                float wf = wreg[ct][r] * gta[r] * scale;
                wreg[ct][r] = wf; oa[r] = wf;
            }
            #pragma unroll
            for (int r = 0; r < 4; ++r) {
                float wf = wreg[ct][r + 4] * gtb[r] * scale;
                wreg[ct][r + 4] = wf; ob[r] = wf;
            }
            *(f32x4*)(wrow + ct * 16)     = oa;
            *(f32x4*)(wrow + ct * 16 + 4) = ob;
        }
    }
    __syncthreads();

    // ---------------- Phase 0c: prototypes -> LDS (d-major bf16) ----------------
    for (int i = tid; i < NC * DIM; i += 256) {
        int c = i >> 7;          // i / 128
        int d = i & (DIM - 1);   // i % 128
        smat[d * STRIDE_DM + c] = (__bf16)protos[i];   // coalesced global read
    }
    __syncthreads();

    // ---------------- Build w^T B-fragments in registers (shfl with partner lane) ----------
    // B frag k, elem j <-> w[myrow][32k + 16*hl + j]; lane holds c%16 in [8*hl, 8*hl+8),
    // the other 8 come from partner lane (^16).
    bf16x16 bw[8];
    #pragma unroll
    for (int k = 0; k < 8; ++k) {
        #pragma unroll
        for (int r = 0; r < 8; ++r) {
            float own_lo = wreg[2 * k][r];       // valid half: lanes 0-15
            float own_hi = wreg[2 * k + 1][r];   // valid half: lanes 16-31
            float sel = hl ? own_lo : own_hi;
            float got = __shfl_xor(sel, 16, 32);
            float vlo = hl ? got : own_lo;
            float vhi = hl ? own_hi : got;
            bw[k][r]     = (__bf16)vlo;
            bw[k][r + 8] = (__bf16)vhi;
        }
    }

    // ---------------- Pass 3: blended^T = P^T * w^T ----------------
    {
        float* brow = out_blended + (size_t)myrow * DIM + 8 * hl;
        const __bf16* abase = smat + ln16 * STRIDE_DM + 16 * hl;
        #pragma unroll
        for (int dt = 0; dt < 8; ++dt) {
            const __bf16* rp = abase + dt * 16 * STRIDE_DM;
            // all 16 fragment loads first, then the 8 WMMAs
            bf16x16 a[8];
            #pragma unroll
            for (int k = 0; k < 8; ++k) {
                bf16x8 lo = *(const bf16x8*)(rp + 32 * k);
                bf16x8 hi = *(const bf16x8*)(rp + 32 * k + 8);
                a[k] = join8(lo, hi);
            }
            f32x8 acc = {0.f, 0.f, 0.f, 0.f, 0.f, 0.f, 0.f, 0.f};
            #pragma unroll
            for (int k = 0; k < 8; ++k)
                acc = __builtin_amdgcn_wmma_f32_16x16x32_bf16(false, a[k], false, bw[k],
                                                              (short)0, acc, false, false);
            f32x4 oa, ob;
            #pragma unroll
            for (int r = 0; r < 4; ++r) { oa[r] = acc[r]; ob[r] = acc[r + 4]; }
            *(f32x4*)(brow + dt * 16)     = oa;
            *(f32x4*)(brow + dt * 16 + 4) = ob;
        }
    }
}

extern "C" void kernel_launch(void* const* d_in, const int* in_sizes, int n_in,
                              void* d_out, int out_size, void* d_ws, size_t ws_size,
                              hipStream_t stream) {
    const float* x      = (const float*)d_in[0];
    const float* protos = (const float*)d_in[1];
    const float* grid   = (const float*)d_in[2];
    const float* traw   = (const float*)d_in[3];
    const float* glog   = (const float*)d_in[4];

    const int rows = in_sizes[0] / DIM;            // 262144
    float* out_b = (float*)d_out;                  // [rows, 128]
    float* out_w = (float*)d_out + (size_t)rows * DIM;  // [rows, 256]

    const int blocks = rows / ROWS_PER_BLOCK;      // 2048
    hipLaunchKernelGGL(softsom_kernel, dim3(blocks), dim3(256), 0, stream,
                       x, protos, grid, traw, glog, out_b, out_w);
}
